// MusicHGT_83829171683607
// MI455X (gfx1250) — compile-verified, matrix operand order checked by reading, not run on hardware
//
#include <hip/hip_runtime.h>
#include <math.h>

// ---------------- model constants ----------------
#define HID 128
#define HEADS 4
#define DH 32
#define LAYERS 3
#define NUM_CLASSES 513
#define GK 128          // K dimension of every dense GEMM in this model
#define GKB (GK / 32)   // WMMA K-blocks

static const int N_OCC = 100000, N_CHORD = 512, N_SEC = 10000, N_NOTE = 128, N_SD = 7;
static const int COUNTS[5] = {100000, 512, 10000, 128, 7};
static const int OFFS[6]   = {0, 100000, 100512, 110512, 110640, 110647};
static const int N_TOT     = 110647;
static const int ESRC[10]  = {0,0,1,0,2,2,1,3,1,4};
static const int EDST[10]  = {0,1,0,2,0,2,3,1,4,1};

typedef __attribute__((ext_vector_type(16))) _Float16 v16h;
typedef __attribute__((ext_vector_type(8)))  float    v8f;

// order-preserving float <-> uint for atomicMax on floats
__device__ __forceinline__ unsigned f32_ord(float f) {
  unsigned u = __float_as_uint(f);
  return (u & 0x80000000u) ? ~u : (u | 0x80000000u);
}
__device__ __forceinline__ float ord_f32(unsigned u) {
  unsigned b = (u & 0x80000000u) ? (u & 0x7FFFFFFFu) : ~u;
  return __uint_as_float(b);
}

// ---------------- weight pack: f32 [GK,N] -> f16 B-fragments ----------------
// Layout: WP[((nt*GKB + kb)*32 + lane)*16 + elem], elem order matches the
// 16-bit B fragment of v_wmma_f32_16x16x32_f16 (mirror of the A 16x32 layout).
__global__ void pack_w_kernel(const float* __restrict__ W, int N,
                              _Float16* __restrict__ WP)
{
  const int Nt = (N + 15) >> 4;
  const int total = Nt * GKB * 32;
  int idx = blockIdx.x * blockDim.x + threadIdx.x;
  if (idx >= total) return;
  const int lane = idx & 31;
  const int kb   = (idx >> 5) & (GKB - 1);
  const int nt   = idx / (32 * GKB);
  int nB = nt * 16 + (lane & 15);
  if (nB >= N) nB = N - 1;                       // clamp; garbage cols never stored
  const int kHalf = (lane >> 4) ? 8 : 0;
  _Float16* dst = WP + (size_t)idx * 16;
#pragma unroll
  for (int j = 0; j < 8; ++j) {
    const int kk = kb * 32 + ((j >= 4) ? 16 : 0) + kHalf + (j & 3) * 2;
    dst[2 * j]     = (_Float16)W[(size_t)kk * N + nB];
    dst[2 * j + 1] = (_Float16)W[(size_t)(kk + 1) * N + nB];
  }
}

// ---------------- WMMA GEMM: C[M,N] = A[M,128] * W[128,N] + bias ------------
// A f32 row-major (lda), W pre-packed f16 fragments, f32 accumulate.
// Register-blocked: one wave computes a 16x64 tile (4 n-tiles share one A
// fragment -> 4x less A traffic). Block = 128 threads = 4 waves = 256 cols.
__global__ void gemm_wmma_packed(const float* __restrict__ A, int lda,
                                 const v16h* __restrict__ WP,
                                 const float* __restrict__ bias,
                                 float* __restrict__ C, int ldc,
                                 int M, int N)
{
  const int wave = threadIdx.x >> 5;
  const int lane = threadIdx.x & 31;
  const int row0 = blockIdx.x * 16;
  const int nt0  = (blockIdx.y * 4 + wave) * 4;   // first of 4 n-tiles
  if (nt0 * 16 >= N) return;             // wave-uniform exit (EXEC stays all-ones)
  const int Nt = (N + 15) >> 4;

  const int mA    = lane & 15;
  const int kHalf = (lane >> 4) ? 8 : 0;
  int mrow = row0 + mA;
  if (mrow >= M) mrow = M - 1;           // clamp; garbage rows never stored
  const float* arow = A + (size_t)mrow * lda;

  const v16h* bp[4];
#pragma unroll
  for (int i = 0; i < 4; ++i) {
    int nti = nt0 + i;
    if (nti >= Nt) nti = Nt - 1;         // clamp; guarded at store
    bp[i] = WP + (size_t)nti * GKB * 32 + lane;
  }

  v8f acc[4] = {v8f{}, v8f{}, v8f{}, v8f{}};
#pragma unroll
  for (int kb = 0; kb < GKB; ++kb) {
    // A fragment: two contiguous 8-float segments per lane -> 4x float4 loads
    const float4* s0 = (const float4*)(arow + kb * 32 + kHalf);
    const float4* s1 = (const float4*)(arow + kb * 32 + 16 + kHalf);
    const float4 p0 = s0[0], p1 = s0[1], p2 = s1[0], p3 = s1[1];
    v16h a;
    a[0]  = (_Float16)p0.x;  a[1]  = (_Float16)p0.y;
    a[2]  = (_Float16)p0.z;  a[3]  = (_Float16)p0.w;
    a[4]  = (_Float16)p1.x;  a[5]  = (_Float16)p1.y;
    a[6]  = (_Float16)p1.z;  a[7]  = (_Float16)p1.w;
    a[8]  = (_Float16)p2.x;  a[9]  = (_Float16)p2.y;
    a[10] = (_Float16)p2.z;  a[11] = (_Float16)p2.w;
    a[12] = (_Float16)p3.x;  a[13] = (_Float16)p3.y;
    a[14] = (_Float16)p3.z;  a[15] = (_Float16)p3.w;
#pragma unroll
    for (int i = 0; i < 4; ++i) {
      const v16h b = bp[i][kb * 32];     // one contiguous 32B fragment load
      acc[i] = __builtin_amdgcn_wmma_f32_16x16x32_f16(false, a, false, b,
                                                      (short)0, acc[i], false, false);
    }
  }

  // C/D layout: VGPR v -> M = v (+8 for lanes 16-31), N = lane & 15
  const int mBase = row0 + ((lane >> 4) ? 8 : 0);
#pragma unroll
  for (int i = 0; i < 4; ++i) {
    const int n = (nt0 + i) * 16 + (lane & 15);
    if (n < N) {
      const float bv = bias ? bias[n] : 0.0f;
#pragma unroll
      for (int v = 0; v < 8; ++v) {
        const int m = mBase + v;
        if (m < M) C[(size_t)m * ldc + n] = acc[i][v] + bv;
      }
    }
  }
}

// ---------------- occ -> chord map (cfpo scatter) ----------------
__global__ void build_map_kernel(const int* __restrict__ ei2, int E, int* __restrict__ map) {
  int i = blockIdx.x * blockDim.x + threadIdx.x;
  if (i < E) map[ei2[E + i]] = ei2[i];   // map[dst_occ] = src_chord
}

// ---------------- input projections (tiny K, VALU) ----------------
__global__ void proj_occ_kernel(const float* __restrict__ xo, const float* __restrict__ xc,
                                const int* __restrict__ map,
                                const float* __restrict__ W, const float* __restrict__ b,
                                float* __restrict__ Hout, int N)
{
  int idx = blockIdx.x * blockDim.x + threadIdx.x;
  if (idx >= N * HID) return;
  int n = idx >> 7, j = idx & 127;
  float s = b[j];
  const float* xr = xo + (size_t)n * 16;
#pragma unroll
  for (int k = 0; k < 16; ++k) s += xr[k] * W[k * HID + j];
  int c = map[n];
  if (c >= 0) {
    const float* cr = xc + (size_t)c * 32;
#pragma unroll
    for (int k = 0; k < 32; ++k) s += cr[k] * W[(16 + k) * HID + j];
  }
  Hout[(size_t)n * HID + j] = s;
}

__global__ void proj_generic_kernel(const float* __restrict__ X, int F,
                                    const float* __restrict__ W, const float* __restrict__ b,
                                    float* __restrict__ Hout, int N)
{
  int idx = blockIdx.x * blockDim.x + threadIdx.x;
  if (idx >= N * HID) return;
  int n = idx >> 7, j = idx & 127;
  float s = b[j];
  const float* xr = X + (size_t)n * F;
  for (int k = 0; k < F; ++k) s += xr[k] * W[k * HID + j];
  Hout[(size_t)n * HID + j] = s;
}

// ---------------- edge pass 1: alpha + segment max ----------------
// 256 threads = 8 waves = 8 edges per block; k_rel staged in LDS (16KB).
__global__ void edge_alpha_kernel(const float* __restrict__ KQV,
                                  const int* __restrict__ ei, int E,
                                  int offS, int offD,
                                  const float* __restrict__ k_rel,
                                  const float* __restrict__ p_rel,
                                  float* __restrict__ aedge,
                                  unsigned* __restrict__ amax)
{
  __shared__ float srel[HEADS * DH * DH];
  for (int i = threadIdx.x; i < HEADS * DH * DH; i += 256) srel[i] = k_rel[i];
  __syncthreads();

  const int wave = threadIdx.x >> 5, lane = threadIdx.x & 31;
  const int e = blockIdx.x * 8 + wave;
  if (e >= E) return;
  const int src = ei[e], dst = ei[E + e];
  const float* krow = KQV + (size_t)(offS + src) * 384;        // k of src
  const float* qrow = KQV + (size_t)(offD + dst) * 384 + 128;  // q of dst
  const int h  = lane >> 3;
  const int lw = lane & 7;
  float s = 0.0f;
#pragma unroll
  for (int t = 0; t < 4; ++t) {
    const int ec = lw * 4 + t;
    const float* kr = srel + h * DH * DH + ec;  // k_rel[h, d, ec]
    const float* kh = krow + h * DH;
    float ke = 0.0f;
#pragma unroll
    for (int d = 0; d < DH; ++d) ke += kh[d] * kr[d * DH];
    s += qrow[h * DH + ec] * ke;
  }
  s += __shfl_xor(s, 1, 32);
  s += __shfl_xor(s, 2, 32);
  s += __shfl_xor(s, 4, 32);
  if (lw == 0) {
    const float a = s * p_rel[h] * 0.17677669529663687f;  // 1/sqrt(32)
    aedge[(size_t)e * HEADS + h] = a;
    atomicMax(&amax[(size_t)(offD + dst) * HEADS + h], f32_ord(a));
  }
}

// ---------------- edge pass 2: exp + segment sum ----------------
__global__ void edge_exp_kernel(const int* __restrict__ ei, int E, int offD,
                                float* __restrict__ aedge,
                                const unsigned* __restrict__ amax,
                                float* __restrict__ den)
{
  int i = blockIdx.x * blockDim.x + threadIdx.x;
  if (i >= E * HEADS) return;
  int e = i >> 2, h = i & 3;
  int dg = offD + ei[E + e];
  float m  = ord_f32(amax[(size_t)dg * HEADS + h]);
  float ex = __expf(aedge[i] - m);
  aedge[i] = ex;
  atomicAdd(&den[(size_t)dg * HEADS + h], ex);
}

// ---------------- edge pass 3: weighted v_rel message scatter ----------------
// 8 edges per 256-thread block; v_rel staged in LDS; lane -> 4 output dims.
__global__ void edge_aggr_kernel(const float* __restrict__ KQV,
                                 const int* __restrict__ ei, int E,
                                 int offS, int offD,
                                 const float* __restrict__ v_rel,
                                 const float* __restrict__ aedge,
                                 const float* __restrict__ den,
                                 float* __restrict__ agg)
{
  __shared__ float srel[HEADS * DH * DH];
  for (int i = threadIdx.x; i < HEADS * DH * DH; i += 256) srel[i] = v_rel[i];
  __syncthreads();

  const int wave = threadIdx.x >> 5, lane = threadIdx.x & 31;
  const int e = blockIdx.x * 8 + wave;
  if (e >= E) return;
  const int src = ei[e];
  const int dg  = offD + ei[E + e];
  const float* vrow = KQV + (size_t)(offS + src) * 384 + 256;  // v of src
#pragma unroll
  for (int t = 0; t < 4; ++t) {
    const int dim = lane * 4 + t;
    const int h = dim >> 5, ec = dim & 31;
    const float* vr = srel + h * DH * DH + ec;
    const float* vh = vrow + h * DH;
    float ve = 0.0f;
#pragma unroll
    for (int d = 0; d < DH; ++d) ve += vh[d] * vr[d * DH];
    const float w = aedge[(size_t)e * HEADS + h] / den[(size_t)dg * HEADS + h];
    atomicAdd(&agg[(size_t)dg * HID + dim], w * ve);
  }
}

// ---------------- exact GELU (in place) ----------------
__global__ void gelu_kernel(float* __restrict__ X, long n) {
  long i = (long)blockIdx.x * blockDim.x + threadIdx.x;
  if (i < n) {
    float x = X[i];
    X[i] = 0.5f * x * (1.0f + erff(x * 0.7071067811865475f));
  }
}

// ---------------- sigmoid-skip blend + LayerNorm + residual (in place) -------
__global__ void blend_ln_res_kernel(const float* __restrict__ O, float* __restrict__ H,
                                    const float* __restrict__ skip,
                                    const float* __restrict__ g, const float* __restrict__ b,
                                    int N)
{
  const int wave = threadIdx.x >> 5, lane = threadIdx.x & 31;
  const int n = blockIdx.x * 4 + wave;
  if (n >= N) return;
  const float s = 1.0f / (1.0f + __expf(-skip[0]));
  const float* orow = O + (size_t)n * HID;
  float* hrow = H + (size_t)n * HID;
  float x[4], h0[4];
  float sum = 0.0f;
#pragma unroll
  for (int t = 0; t < 4; ++t) {
    int j = lane * 4 + t;
    h0[t] = hrow[j];
    x[t]  = s * orow[j] + (1.0f - s) * h0[t];
    sum  += x[t];
  }
  for (int m = 16; m >= 1; m >>= 1) sum += __shfl_xor(sum, m, 32);
  const float mean = sum * (1.0f / HID);
  float var = 0.0f;
#pragma unroll
  for (int t = 0; t < 4; ++t) { float d = x[t] - mean; var += d * d; }
  for (int m = 16; m >= 1; m >>= 1) var += __shfl_xor(var, m, 32);
  const float rs = rsqrtf(var * (1.0f / HID) + 1e-5f);
#pragma unroll
  for (int t = 0; t < 4; ++t) {
    int j = lane * 4 + t;
    hrow[j] = (x[t] - mean) * rs * g[j] + b[j] + h0[t];
  }
}

// ---------------- host driver ----------------
static inline unsigned cdiv(long a, long b) { return (unsigned)((a + b - 1) / b); }

static void launch_pack(const float* W, int N, _Float16* WP, hipStream_t s) {
  int total = ((N + 15) >> 4) * GKB * 32;
  pack_w_kernel<<<cdiv(total, 256), 256, 0, s>>>(W, N, WP);
}
static void launch_gemm(const float* A, int lda, const _Float16* WP, const float* bias,
                        float* C, int ldc, int M, int N, hipStream_t s)
{
  dim3 grid(cdiv(M, 16), cdiv(N, 256));   // wave = 16x64, block = 16x256
  gemm_wmma_packed<<<grid, dim3(128), 0, s>>>(A, lda, (const v16h*)WP, bias, C, ldc, M, N);
}

extern "C" void kernel_launch(void* const* d_in, const int* in_sizes, int n_in,
                              void* d_out, int out_size, void* d_ws, size_t ws_size,
                              hipStream_t stream)
{
  // ---- inputs (setup_inputs dict order) ----
  const float* x_occ   = (const float*)d_in[0];
  const float* x_chord = (const float*)d_in[1];
  const float* x_sec   = (const float*)d_in[2];
  const float* x_note  = (const float*)d_in[3];
  const float* x_sd    = (const float*)d_in[4];
  const float* pw_occ   = (const float*)d_in[5];
  const float* pb_occ   = (const float*)d_in[6];
  const float* pw_chord = (const float*)d_in[7];
  const float* pb_chord = (const float*)d_in[8];
  const float* pw_sec   = (const float*)d_in[9];
  const float* pb_sec   = (const float*)d_in[10];
  const float* pw_note  = (const float*)d_in[11];
  const float* pb_note  = (const float*)d_in[12];
  const float* pw_sd    = (const float*)d_in[13];
  const float* pb_sd    = (const float*)d_in[14];
  const float* kqv_w = (const float*)d_in[15];  // [3,5,128,384]
  const float* kqv_b = (const float*)d_in[16];  // [3,5,384]
  const float* k_rel = (const float*)d_in[17];  // [3,10,4,32,32]
  const float* v_rel = (const float*)d_in[18];  // [3,10,4,32,32]
  const float* p_rel = (const float*)d_in[19];  // [3,10,4]
  const float* out_w = (const float*)d_in[20];  // [3,5,128,128]
  const float* out_b = (const float*)d_in[21];  // [3,5,128]
  const float* skip  = (const float*)d_in[22];  // [3,5]
  const float* ln_g  = (const float*)d_in[23];  // [3,128]
  const float* ln_b  = (const float*)d_in[24];  // [3,128]
  const float* cls_w = (const float*)d_in[25];  // [128,513]
  const float* cls_b = (const float*)d_in[26];  // [513]

  const int* ei[10]; int E[10]; long eoff[10], Etot = 0;
  for (int e = 0; e < 10; ++e) {
    ei[e]   = (const int*)d_in[27 + e];
    E[e]    = in_sizes[27 + e] / 2;
    eoff[e] = Etot;
    Etot   += E[e];
  }

  // ---- workspace carve (256B aligned) ----
  char* p = (char*)d_ws;
  auto carve = [&](size_t bytes) -> char* {
    char* r = p;
    p += (bytes + 255) & ~(size_t)255;
    return r;
  };
  float*    H     = (float*)   carve((size_t)N_TOT * HID * 4);
  float*    KQV   = (float*)   carve((size_t)N_TOT * 384 * 4);
  float*    AGG   = (float*)   carve((size_t)N_TOT * HID * 4);
  float*    HN    = (float*)   carve((size_t)N_TOT * HID * 4);
  unsigned* AMAX  = (unsigned*)carve((size_t)N_TOT * HEADS * 4);
  float*    DEN   = (float*)   carve((size_t)N_TOT * HEADS * 4);
  float*    AEDGE = (float*)   carve((size_t)Etot * HEADS * 4);
  int*      MAP   = (int*)     carve((size_t)N_OCC * 4);
  // packed f16 weights
  const size_t KQV_FR = (size_t)(384 / 16) * GKB * 32 * 16;       // halves per (l,t)
  const size_t OUT_FR = (size_t)(HID / 16) * GKB * 32 * 16;
  const size_t CLS_FR = (size_t)((NUM_CLASSES + 15) / 16) * GKB * 32 * 16;
  _Float16* WP_KQV = (_Float16*)carve(15 * KQV_FR * 2);
  _Float16* WP_OUT = (_Float16*)carve(15 * OUT_FR * 2);
  _Float16* WP_CLS = (_Float16*)carve(CLS_FR * 2);
  (void)ws_size; (void)n_in;

  // ---- pack all GEMM weights to f16 fragments (once per launch) ----
  for (int i = 0; i < 15; ++i) {
    launch_pack(kqv_w + (size_t)i * HID * 384, 384, WP_KQV + (size_t)i * KQV_FR, stream);
    launch_pack(out_w + (size_t)i * HID * HID, HID, WP_OUT + (size_t)i * OUT_FR, stream);
  }
  launch_pack(cls_w, NUM_CLASSES, WP_CLS, stream);

  // ---- build h (input projections) ----
  hipMemsetAsync(MAP, 0xFF, (size_t)N_OCC * 4, stream);  // -1
  build_map_kernel<<<cdiv(E[2], 256), 256, 0, stream>>>(ei[2], E[2], MAP);

  proj_occ_kernel<<<cdiv((long)N_OCC * HID, 256), 256, 0, stream>>>(
      x_occ, x_chord, MAP, pw_occ, pb_occ, H + (size_t)OFFS[0] * HID, N_OCC);
  proj_generic_kernel<<<cdiv((long)N_CHORD * HID, 256), 256, 0, stream>>>(
      x_chord, 32, pw_chord, pb_chord, H + (size_t)OFFS[1] * HID, N_CHORD);
  proj_generic_kernel<<<cdiv((long)N_SEC * HID, 256), 256, 0, stream>>>(
      x_sec, 8, pw_sec, pb_sec, H + (size_t)OFFS[2] * HID, N_SEC);
  proj_generic_kernel<<<cdiv((long)N_NOTE * HID, 256), 256, 0, stream>>>(
      x_note, 12, pw_note, pb_note, H + (size_t)OFFS[3] * HID, N_NOTE);
  proj_generic_kernel<<<cdiv((long)N_SD * HID, 256), 256, 0, stream>>>(
      x_sd, 7, pw_sd, pb_sd, H + (size_t)OFFS[4] * HID, N_SD);

  // ---- HGT layers ----
  for (int l = 0; l < LAYERS; ++l) {
    for (int t = 0; t < 5; ++t) {
      launch_gemm(H + (size_t)OFFS[t] * HID, HID,
                  WP_KQV + (size_t)(l * 5 + t) * KQV_FR,
                  kqv_b + (size_t)(l * 5 + t) * 384,
                  KQV + (size_t)OFFS[t] * 384, 384,
                  COUNTS[t], 384, stream);
    }

    hipMemsetAsync(AMAX, 0x00, (size_t)N_TOT * HEADS * 4, stream);  // ordered -inf
    hipMemsetAsync(DEN,  0x00, (size_t)N_TOT * HEADS * 4, stream);
    hipMemsetAsync(AGG,  0x00, (size_t)N_TOT * HID * 4, stream);

    for (int e = 0; e < 10; ++e) {
      edge_alpha_kernel<<<cdiv(E[e], 8), 256, 0, stream>>>(
          KQV, ei[e], E[e], OFFS[ESRC[e]], OFFS[EDST[e]],
          k_rel + (size_t)(l * 10 + e) * HEADS * DH * DH,
          p_rel + (size_t)(l * 10 + e) * HEADS,
          AEDGE + eoff[e] * HEADS, AMAX);
    }
    for (int e = 0; e < 10; ++e) {
      edge_exp_kernel<<<cdiv((long)E[e] * HEADS, 256), 256, 0, stream>>>(
          ei[e], E[e], OFFS[EDST[e]], AEDGE + eoff[e] * HEADS, AMAX, DEN);
    }
    for (int e = 0; e < 10; ++e) {
      edge_aggr_kernel<<<cdiv(E[e], 8), 256, 0, stream>>>(
          KQV, ei[e], E[e], OFFS[ESRC[e]], OFFS[EDST[e]],
          v_rel + (size_t)(l * 10 + e) * HEADS * DH * DH,
          AEDGE + eoff[e] * HEADS, DEN, AGG);
    }

    gelu_kernel<<<cdiv((long)N_TOT * HID, 256), 256, 0, stream>>>(AGG, (long)N_TOT * HID);

    for (int t = 0; t < 5; ++t) {
      launch_gemm(AGG + (size_t)OFFS[t] * HID, HID,
                  WP_OUT + (size_t)(l * 5 + t) * OUT_FR,
                  out_b + (size_t)(l * 5 + t) * HID,
                  HN + (size_t)OFFS[t] * HID, HID,
                  COUNTS[t], HID, stream);
    }
    for (int t = 0; t < 5; ++t) {
      blend_ln_res_kernel<<<cdiv(COUNTS[t], 4), 128, 0, stream>>>(
          HN + (size_t)OFFS[t] * HID, H + (size_t)OFFS[t] * HID,
          skip + (size_t)(l * 5 + t),
          ln_g + (size_t)l * HID, ln_b + (size_t)l * HID, COUNTS[t]);
    }
  }

  // ---- classifier: logits = H_occ @ cls_w + cls_b ----
  launch_gemm(H, HID, WP_CLS, cls_b, (float*)d_out, NUM_CLASSES,
              N_OCC, NUM_CLASSES, stream);
  (void)out_size;
}
